// NONA_35562329211596
// MI455X (gfx1250) — matrix operand order repaired and checked by reading, not compiled, not run on hardware
//
#include <hip/hip_runtime.h>
#include <hip/hip_bf16.h>

// ---------------------------------------------------------------------------
// Fused soft-NN regression for MI455X (gfx1250, wave32, WMMA + TDM).
//   sim[i,j] = 2*x_i.xn_j - ||x_i||^2 - ||xn_j||^2 ; softmax_j ; @ y_n ; clip
// -||x_i||^2 cancels in the softmax; only xn2[j] needed.
// Flash-style online softmax with PER-LANE state (merged once at the end).
// fp32 inputs split into bf16 hi/lo; per K-step hi*hi + hi*lo + lo*hi with
// v_wmma_f32_16x16x32_bf16 (fp32 accumulate), sub-tile-interleaved so
// consecutive WMMAs hit different accumulators (hides XDL latency).
// B tiles staged to LDS by the Tensor Data Mover (tensor_load_to_lds,
// TENSORcnt), one issue per tile from wave 0, double-buffered; D# uses
// pad_enable to give LDS rows a 528B stride -> bank-conflict-free
// ds_load_b128. Fallback: global_load_async_to_lds_b128 (ASYNCcnt).
// 4 waves/block share each B tile -> 4x less L2 traffic; M split 4 ways
// across blockIdx.y; partial (m,l,acc) merged by an epilogue kernel.
// ---------------------------------------------------------------------------

typedef __attribute__((ext_vector_type(16))) __bf16 v16bf;
typedef __attribute__((ext_vector_type(8)))  __bf16 v8bf;
typedef __attribute__((ext_vector_type(8)))  float  v8f;
typedef __attribute__((ext_vector_type(4)))  unsigned int v4ui;
typedef __attribute__((ext_vector_type(8)))  int v8i;
typedef __attribute__((ext_vector_type(4)))  int v4i;
typedef __attribute__((ext_vector_type(4)))  unsigned int v4u;

typedef __attribute__((address_space(1))) v4i* gas1_p;   // global int4*
typedef __attribute__((address_space(3))) v4i* las3_p;   // LDS int4*
typedef __attribute__((address_space(3))) void* las3_vp;

#define DK      128   // feature dimension (fixed by the reference)
#define MSPLIT  4     // split of the M (column) loop across blockIdx.y
#define TCOLS   64    // columns per staged tile
#define RSTRIDE 528   // LDS row stride: 512B data + 16B pad (4i mod 64 banks)
#define BUFSZ   (TCOLS * RSTRIDE)   // 33792 B per buffer

#if __has_builtin(__builtin_amdgcn_tensor_load_to_lds)
#define USE_TDM 1
#endif
#if !defined(USE_TDM) && __has_builtin(__builtin_amdgcn_global_load_async_to_lds_b128)
#define USE_ASYNC 1
#endif

union AV { v16bf v; v8bf h[2]; };

__device__ __forceinline__ unsigned int lds_off(void* p) {
    return (unsigned int)(size_t)(las3_vp)p;
}

#ifdef USE_TDM
// One TDM issue loads a whole 64-row x 512B tile (hi|lo interleaved rows)
// into LDS with a 16B pad after every 512B row (LDS stride 528B).
// D# per CDNA5 ISA ch.8: group0 = {count, lds_addr, global_addr, type=2};
// group1 = {data_size=4B, pad_interval=128dw, pad_amount=4dw, dims/strides}.
__device__ __forceinline__ void tdm_load_tile(const void* gsrc, unsigned int ldsoff) {
    unsigned long long ga = (unsigned long long)(size_t)gsrc;
    v4ui g0 = { 1u,                                  // count=1, user mode
                ldsoff,                              // lds_addr
                (unsigned int)ga,                    // global_addr[31:0]
                (unsigned int)((ga >> 32) & 0x01FFFFFFu) | (2u << 30) }; // [56:32] | type=2
    v8i  g1 = { (int)((2u << 16) | (1u << 20) | (6u << 22) | (3u << 25)),
                                                     // data_size=4B, pad_en,
                                                     // interval=128dw, amount=4dw
                (int)(128u << 16),                   // tensor_dim0 = 128 (x4B = 512B row)
                (int)(64u  << 16),                   // tensor_dim0 hi=0 | tensor_dim1 = 64
                (int)(128u << 16),                   // tensor_dim1 hi=0 | tile_dim0 = 128
                64,                                  // tile_dim1 = 64 | tile_dim2 = 0
                128,                                 // tensor_dim0_stride = 128 (rows contiguous)
                0, 0 };
    v4i  z4 = { 0, 0, 0, 0 };
#if __clang_major__ >= 23
    v8i  z8 = { 0, 0, 0, 0, 0, 0, 0, 0 };
    __builtin_amdgcn_tensor_load_to_lds(g0, g1, z4, z4, z8, 0);  // 6-arg form
#else
    __builtin_amdgcn_tensor_load_to_lds(g0, g1, z4, z4, 0);      // 5-arg form
#endif
}
#else
__device__ __forceinline__ void cp_b128(void* lds, const void* g) {
#ifdef USE_ASYNC
    __builtin_amdgcn_global_load_async_to_lds_b128((gas1_p)g, (las3_p)lds, 0, 0);
#else
    *(v4u*)lds = *(const v4u*)g;
#endif
}
// All 128 threads cooperatively stage one tile (2048 x 16B chunks).
__device__ __forceinline__ void stage_async(unsigned char* buf,
                                            const unsigned char* g, int tid) {
    #pragma unroll
    for (int rr = 0; rr < 16; ++rr) {
        int n = rr * 128 + tid;
        int row = n >> 5, c = n & 31;
        cp_b128(buf + row * RSTRIDE + c * 16, g + (size_t)n * 16);
    }
}
#endif

// Publish the staged buffer: producer waits its DMA counter, then barrier.
__device__ __forceinline__ void publish_barrier(int wave) {
#if defined(USE_TDM)
    if (wave == 0) __builtin_amdgcn_s_wait_tensorcnt(0);
#elif defined(USE_ASYNC)
#if __has_builtin(__builtin_amdgcn_s_wait_asynccnt)
    __builtin_amdgcn_s_wait_asynccnt(0);
#else
    asm volatile("s_wait_asynccnt 0x0" ::: "memory");
#endif
#endif
    __syncthreads();
}

// --- Prep 1: split fp32 -> bf16 hi/lo; x separate, x_n row-interleaved ------
__global__ void NONA_prep_split(const float* __restrict__ x,
                                const float* __restrict__ xn,
                                unsigned short* __restrict__ xhi_u,
                                unsigned short* __restrict__ xlo_u,
                                unsigned short* __restrict__ xnhl_u,  // [M][hi 128 | lo 128]
                                int n_x, int n_xn, int* mism, int same)
{
    __bf16* xhi  = (__bf16*)xhi_u;
    __bf16* xlo  = (__bf16*)xlo_u;
    __bf16* xnhl = (__bf16*)xnhl_u;

    int i = blockIdx.x * blockDim.x + threadIdx.x;
    __shared__ int s_any;
    if (threadIdx.x == 0) s_any = 0;
    __syncthreads();

    bool diff = false;
    if (i < n_x) {
        float v = x[i];
        __bf16 h = (__bf16)v;
        xhi[i] = h;
        xlo[i] = (__bf16)(v - (float)h);
    }
    if (i < n_xn) {
        float v = xn[i];
        __bf16 h = (__bf16)v;
        int row = i >> 7, k = i & (DK - 1);           // DK == 128
        xnhl[(size_t)row * 256 + k]       = h;
        xnhl[(size_t)row * 256 + 128 + k] = (__bf16)(v - (float)h);
        if (same && v != x[i]) diff = true;           // same==1 => n_x == n_xn
    }
    if (diff) s_any = 1;                              // benign race
    __syncthreads();
    if (threadIdx.x == 0 && s_any) atomicOr(mism, 1);
}

// --- Prep 2: exact fp32 row norms of x_n (one wave per row) -----------------
__global__ void NONA_rownorm(const float* __restrict__ xn,
                             float* __restrict__ xn2, int M, int D)
{
    int lane = threadIdx.x & 31;
    int wave = threadIdx.x >> 5;
    int row  = blockIdx.x * (blockDim.x >> 5) + wave;
    if (row >= M) return;
    const float* p = xn + (size_t)row * D;
    float s = 0.f;
    for (int k = lane; k < D; k += 32) { float v = p[k]; s += v * v; }
    #pragma unroll
    for (int off = 16; off > 0; off >>= 1) s += __shfl_xor(s, off, 32);
    if (lane == 0) xn2[row] = s;
}

// --- Main: flash softmax-NN with WMMA + TDM-staged B ------------------------
// Block = 128 threads = 4 waves; each wave owns a 16-row strip (block: 64
// rows); all 4 waves share each B tile through LDS. blockIdx.y = M split.
// Requires N % 64 == 0 and (M/MSPLIT) % 64 == 0 (true here: 8192).
__global__ void __launch_bounds__(128)
NONA_nnsoftmax(const unsigned short* __restrict__ xhi_u,
               const unsigned short* __restrict__ xlo_u,
               const unsigned short* __restrict__ xnhl_u,
               const float* __restrict__ xn2,
               const float* __restrict__ yn,
               const int* __restrict__ mism,
               float* __restrict__ pm,   // [MSPLIT*N] partial row max
               float* __restrict__ pl,   // [MSPLIT*N] partial row sum
               float* __restrict__ pa,   // [MSPLIT*N] partial row acc
               int N, int M)
{
    __shared__ __align__(16) unsigned char smem[2][BUFSZ];

    const __bf16* xhi = (const __bf16*)xhi_u;
    const __bf16* xlo = (const __bf16*)xlo_u;
    const unsigned char* gxn = (const unsigned char*)xnhl_u;  // 512B rows

    const int tid  = threadIdx.x;
    const int lane = tid & 31;
    const int wave = tid >> 5;
    const int half = lane >> 4;       // 0: lanes 0-15, 1: lanes 16-31
    const int l15  = lane & 15;
    const int row0 = blockIdx.x * 64 + wave * 16;
    const int mper = M / MSPLIT;
    const int jbeg = blockIdx.y * mper;
    const int ntiles = mper / TCOLS;

    const bool is_train = (*mism == 0);

    // A strip (16 rows x 128 K, hi+lo) in registers for the whole kernel.
    // ISA A-layout (16-bit 16x32): lanes 0-15 hold K=c..c+7 & c+16..c+23,
    // lanes 16-31 hold K=c+8..c+15 & c+24..c+31; row M = lane&15.
    AV ah[4], al[4];
    {
        const __bf16* ph  = xhi + (size_t)(row0 + l15) * DK;
        const __bf16* plo = xlo + (size_t)(row0 + l15) * DK;
        #pragma unroll
        for (int t = 0; t < 4; ++t) {
            int c0 = t * 32 + half * 8;
            ah[t].h[0] = *(const v8bf*)(ph + c0);
            ah[t].h[1] = *(const v8bf*)(ph + c0 + 16);
            al[t].h[0] = *(const v8bf*)(plo + c0);
            al[t].h[1] = *(const v8bf*)(plo + c0 + 16);
        }
    }

    // Per-LANE online-softmax state: 8 rows/lane (C-layout row = r + 8*half),
    // each lane only over columns j == (lane&15) mod 16. Merged at the end.
    float m[8], l[8], acc[8];
    #pragma unroll
    for (int r = 0; r < 8; ++r) {
        m[r] = -__builtin_inff(); l[r] = 0.f; acc[r] = 0.f;
    }

    const unsigned int soff0 = lds_off(&smem[0][0]);
    const unsigned int soff1 = lds_off(&smem[1][0]);

    // Prologue: stage tile 0 into buffer 0.
#ifdef USE_TDM
    if (wave == 0) tdm_load_tile(gxn + (size_t)jbeg * 512, soff0);
#else
    stage_async(&smem[0][0], gxn + (size_t)jbeg * 512, tid);
#endif
    (void)soff1;

    const v8f vzero = {0.f, 0.f, 0.f, 0.f, 0.f, 0.f, 0.f, 0.f};

    for (int it = 0; it < ntiles; ++it) {
        const int j0 = jbeg + it * TCOLS;

        // Publishes buffer it&1; also fences last iteration's readers of
        // buffer (it+1)&1 before it is overwritten below.
        publish_barrier(wave);

        if (it + 1 < ntiles) {
            const unsigned char* gn = gxn + (size_t)(j0 + TCOLS) * 512;
#ifdef USE_TDM
            if (wave == 0) tdm_load_tile(gn, (it & 1) ? soff0 : soff1);
#else
            stage_async(&smem[(it + 1) & 1][0], gn, tid);
#endif
        }

        const unsigned char* bb = &smem[it & 1][0];

        v8f   S[4];
        float xn2v[4], yv[4];
        int   jj[4];

        #pragma unroll
        for (int sub = 0; sub < 4; ++sub) {
            const int j = j0 + sub * 16 + l15;   // B column owned by this lane
            jj[sub]   = j;
            xn2v[sub] = xn2[j];
            yv[sub]   = yn[j];
            S[sub]    = vzero;
        }

        // t-major, sub-interleaved: consecutive WMMAs target different
        // accumulators -> 4 independent XDL chains.
        #pragma unroll
        for (int t = 0; t < 4; ++t) {
            AV bh[4], bl[4];
            #pragma unroll
            for (int sub = 0; sub < 4; ++sub) {
                const int rb = (sub * 16 + l15) * RSTRIDE + (t * 4 + half) * 16;
                bh[sub].h[0] = *(const v8bf*)(bb + rb);
                bh[sub].h[1] = *(const v8bf*)(bb + rb + 32);
                bl[sub].h[0] = *(const v8bf*)(bb + rb + 256);
                bl[sub].h[1] = *(const v8bf*)(bb + rb + 256 + 32);
            }
            #pragma unroll
            for (int sub = 0; sub < 4; ++sub)
                S[sub] = __builtin_amdgcn_wmma_f32_16x16x32_bf16(
                             false, ah[t].v, false, bh[sub].v, (short)0, S[sub], false, false);
            #pragma unroll
            for (int sub = 0; sub < 4; ++sub)
                S[sub] = __builtin_amdgcn_wmma_f32_16x16x32_bf16(
                             false, ah[t].v, false, bl[sub].v, (short)0, S[sub], false, false);
            #pragma unroll
            for (int sub = 0; sub < 4; ++sub)
                S[sub] = __builtin_amdgcn_wmma_f32_16x16x32_bf16(
                             false, al[t].v, false, bh[sub].v, (short)0, S[sub], false, false);
        }

        // Per-lane online softmax update (no cross-lane traffic in the loop).
        #pragma unroll
        for (int r = 0; r < 8; ++r) {
            const int row = row0 + r + half * 8;
            float s0 = 2.f * S[0][r] - xn2v[0];
            float s1 = 2.f * S[1][r] - xn2v[1];
            float s2 = 2.f * S[2][r] - xn2v[2];
            float s3 = 2.f * S[3][r] - xn2v[3];
            if (is_train) {                     // mask self-similarity diagonal
                if (row == jj[0]) s0 = -__builtin_inff();
                if (row == jj[1]) s1 = -__builtin_inff();
                if (row == jj[2]) s2 = -__builtin_inff();
                if (row == jj[3]) s3 = -__builtin_inff();
            }
            float tmax = fmaxf(fmaxf(s0, s1), fmaxf(s2, s3));
            float nm   = fmaxf(m[r], tmax);
            float sc   = __expf(m[r] - nm);     // 0 when m was -inf
            float p0 = __expf(s0 - nm);
            float p1 = __expf(s1 - nm);
            float p2 = __expf(s2 - nm);
            float p3 = __expf(s3 - nm);
            l[r]   = l[r]   * sc + (p0 + p1 + p2 + p3);
            acc[r] = acc[r] * sc + (p0 * yv[0] + p1 * yv[1] + p2 * yv[2] + p3 * yv[3]);
            m[r]   = nm;
        }
    }

    // One-time 16-lane softmax-state merge (butterfly within each half-wave),
    // then lane 0 / lane 16 write the per-split partials.
    #pragma unroll
    for (int r = 0; r < 8; ++r) {
        float mm = m[r], ll = l[r], aa = acc[r];
        #pragma unroll
        for (int off = 1; off < 16; off <<= 1) {
            float om = __shfl_xor(mm, off, 32);
            float ol = __shfl_xor(ll, off, 32);
            float oa = __shfl_xor(aa, off, 32);
            float nm = fmaxf(mm, om);
            float c0 = __expf(mm - nm);
            float c1 = __expf(om - nm);
            ll = ll * c0 + ol * c1;
            aa = aa * c0 + oa * c1;
            mm = nm;
        }
        if (l15 == 0) {
            const int row = row0 + r + half * 8;
            const size_t idx = (size_t)blockIdx.y * N + row;
            pm[idx] = mm;
            pl[idx] = ll;
            pa[idx] = aa;
        }
    }
}

// --- Epilogue: merge the MSPLIT partial softmax states per row --------------
__global__ void NONA_merge(const float* __restrict__ pm,
                           const float* __restrict__ pl,
                           const float* __restrict__ pa,
                           float* __restrict__ out, int N)
{
    int row = blockIdx.x * blockDim.x + threadIdx.x;
    if (row >= N) return;
    float m = -__builtin_inff(), l = 0.f, a = 0.f;
    #pragma unroll
    for (int s = 0; s < MSPLIT; ++s) {
        float ms = pm[(size_t)s * N + row];
        float nm = fmaxf(m, ms);
        float c0 = __expf(m - nm);
        float c1 = __expf(ms - nm);
        l = l * c0 + pl[(size_t)s * N + row] * c1;
        a = a * c0 + pa[(size_t)s * N + row] * c1;
        m = nm;
    }
    float v = a / l;
    out[row] = fminf(fmaxf(v, 0.f), 1.f);
}

// ---------------------------------------------------------------------------
extern "C" void kernel_launch(void* const* d_in, const int* in_sizes, int n_in,
                              void* d_out, int out_size, void* d_ws, size_t ws_size,
                              hipStream_t stream) {
    const float* x  = (const float*)d_in[0];
    const float* xn = (const float*)d_in[1];
    const float* yn = (const float*)d_in[2];
    float* out = (float*)d_out;

    const int n_x  = in_sizes[0];       // N*D
    const int n_xn = in_sizes[1];       // M*D
    const int M    = in_sizes[2];
    const int N    = out_size;
    const int D    = n_x / N;           // == 128 for this problem
    const int same = (n_x == n_xn) ? 1 : 0;
    (void)n_in; (void)ws_size;

    // Workspace (~8.8 MB): bf16 hi/lo of x, interleaved hi|lo of x_n, xn2,
    // per-split partials, eq flag.
    unsigned char* w = (unsigned char*)d_ws;
    unsigned short* xhi  = (unsigned short*)w;  w += (size_t)n_x  * 2;
    unsigned short* xlo  = (unsigned short*)w;  w += (size_t)n_x  * 2;
    unsigned short* xnhl = (unsigned short*)w;  w += (size_t)n_xn * 4;  // hi+lo
    float* xn2 = (float*)w;                     w += (size_t)M * 4;
    float* pm  = (float*)w;                     w += (size_t)MSPLIT * N * 4;
    float* pl  = (float*)w;                     w += (size_t)MSPLIT * N * 4;
    float* pa  = (float*)w;                     w += (size_t)MSPLIT * N * 4;
    int*   mism = (int*)w;

    // flag==0 <=> (so far) x == x_n elementwise; size mismatch => not train.
    (void)hipMemsetAsync(mism, same ? 0 : 1, sizeof(int), stream);

    int total = n_x > n_xn ? n_x : n_xn;
    NONA_prep_split<<<(total + 255) / 256, 256, 0, stream>>>(
        x, xn, xhi, xlo, xnhl, n_x, n_xn, mism, same);

    NONA_rownorm<<<(M + 7) / 8, 256, 0, stream>>>(xn, xn2, M, D);

    dim3 grid(N / 64, MSPLIT);
    NONA_nnsoftmax<<<grid, 128, 0, stream>>>(
        xhi, xlo, xnhl, xn2, yn, mism, pm, pl, pa, N, M);

    NONA_merge<<<(N + 255) / 256, 256, 0, stream>>>(pm, pl, pa, out, N);
}
